// HRWKV_7_38534446579912
// MI455X (gfx1250) — matrix operand order repaired and checked by reading, not compiled, not run on hardware
//
#include <hip/hip_runtime.h>

// ---------------------------------------------------------------------------
// RWKV-7 block forward for MI455X (gfx1250, wave32).
// Big projections run as bf16 WMMA GEMMs (v_wmma_f32_16x16x32_bf16, fp32 acc).
// Weights are converted to bf16 *transposed* (NxK) so both A and B fragments
// are K-contiguous per lane -> pure global_load_b128 traffic, no u16 gathers.
// Norms / RoPE / gating / the sequential state scan stay fp32 VALU.
// ---------------------------------------------------------------------------

static constexpr int Hh     = 32;
static constexpr int Nn     = 64;
static constexpr int HKVn   = 8;
static constexpr int HNc    = 2048;   // H*N
static constexpr int KVD    = 512;    // HKV*N
static constexpr int Bc     = 2;
static constexpr int Tc     = 1024;
static constexpr int BT     = Bc * Tc;
static constexpr int WAVGK  = 576;    // 96+96+64+256+64
static constexpr int RKVN   = HNc + 2 * KVD; // 3072

typedef __attribute__((ext_vector_type(16))) __bf16 v16bf;
typedef __attribute__((ext_vector_type(8)))  float  v8f;

union Frag16 { v16bf v; unsigned short u[16]; };

__device__ __forceinline__ unsigned short f32_to_bf16(float f) {
  unsigned int x = __builtin_bit_cast(unsigned int, f);
  unsigned int r = x + 0x7FFFu + ((x >> 16) & 1u);   // round-to-nearest-even
  return (unsigned short)(r >> 16);
}
__device__ __forceinline__ float sigmoidf_(float x) { return 1.0f / (1.0f + expf(-x)); }
__device__ __forceinline__ float softplusf_(float x) {
  if (x > 20.0f)  return x;
  if (x < -20.0f) return expf(x);
  return log1pf(expf(x));
}
__device__ __forceinline__ float block64_sum(float v, float* red, int i) {
  red[i] = v; __syncthreads();
  for (int off = 32; off > 0; off >>= 1) {
    if (i < off) red[i] += red[i + off];
    __syncthreads();
  }
  float s = red[0];
  __syncthreads();
  return s;
}

// ------------------- fp32 (KxN) -> bf16 transposed (NxK) -------------------
// Output-coalesced: flat output index i = n*K + k.
__global__ __launch_bounds__(256) void cvt_transpose_bf16_kernel(
    const float* __restrict__ in, unsigned short* __restrict__ out,
    int K, int N) {
  size_t total = (size_t)K * N;
  size_t i = (size_t)blockIdx.x * 256 + threadIdx.x;
  size_t stride = (size_t)gridDim.x * 256;
  for (; i < total; i += stride) {
    int k = (int)(i % K);
    int n = (int)(i / K);
    out[i] = f32_to_bf16(in[(size_t)k * N + n]);
  }
}

// --------------------------- row RMSNorm -----------------------------------
// out = w * (in+add) * rsqrt(mean((in+add)^2) + eps). Optional fp32/bf16 outs.
__global__ __launch_bounds__(256) void rmsnorm_kernel(
    const float* __restrict__ in, const float* __restrict__ add,
    const float* __restrict__ w, float* __restrict__ out_f32,
    unsigned short* __restrict__ out_bf, int C) {
  __shared__ float red[256];
  const int row = blockIdx.x;
  const float* x = in + (size_t)row * C;
  const float* ax = add ? add + (size_t)row * C : nullptr;
  float s = 0.0f;
  for (int c = threadIdx.x; c < C; c += 256) {
    float v = x[c] + (ax ? ax[c] : 0.0f);
    s += v * v;
  }
  red[threadIdx.x] = s;
  __syncthreads();
  for (int off = 128; off > 0; off >>= 1) {
    if ((int)threadIdx.x < off) red[threadIdx.x] += red[threadIdx.x + off];
    __syncthreads();
  }
  const float scale = rsqrtf(red[0] / (float)C + 1e-6f);
  for (int c = threadIdx.x; c < C; c += 256) {
    float v = (x[c] + (ax ? ax[c] : 0.0f)) * scale * w[c];
    if (out_f32) out_f32[(size_t)row * C + c] = v;
    if (out_bf)  out_bf [(size_t)row * C + c] = f32_to_bf16(v);
  }
}

// --------------------------- bf16 WMMA GEMM --------------------------------
// C[M,N] = A[M,K] * B[K,N] with B supplied TRANSPOSED as Bt[N,K] (bf16).
// Block = 2 waves; each wave owns two adjacent 16x16 C tiles (shared A
// fragment, two B fragments, two independent accumulators -> 2 back-to-back
// WMMAs, no D->A/B hazard). Fragment layouts used here (A/B k order is this code's, not the ISA table's):
//   A 16x32: lane&15 = row, lane>>4 = K-half (k = 16*kh + s, contiguous order); two 8-elem runs.
//   B 32x16: lane&15 = col, lane>>4 = K-half; 16 contiguous K elems (from Bt), same k order as A.
//   C 16x16 f32: VGPR r -> row r + 8*(lane>>4), col = lane&15.
__global__ __launch_bounds__(64) void wmma_gemm_bf16(
    const unsigned short* __restrict__ A, const unsigned short* __restrict__ Bt,
    float* __restrict__ C, int M, int N, int K) {
  const int lane = threadIdx.x & 31;
  const int wave = threadIdx.x >> 5;            // 0..1
  const int col0 = blockIdx.x * 64 + wave * 32; // two 16-col tiles per wave
  const int row0 = blockIdx.y * 16;
  const int m  = lane & 15;
  const int kh = lane >> 4;   // K-half select
  const int n  = lane & 15;

  v8f acc0 = {0.f, 0.f, 0.f, 0.f, 0.f, 0.f, 0.f, 0.f};
  v8f acc1 = {0.f, 0.f, 0.f, 0.f, 0.f, 0.f, 0.f, 0.f};
  const unsigned short* arow  = A  + (size_t)(row0 + m) * K + kh * 16;
  const unsigned short* brow0 = Bt + (size_t)(col0 + n) * K + kh * 16;
  const unsigned short* brow1 = brow0 + (size_t)16 * K;

  for (int k0 = 0; k0 < K; k0 += 32) {
    Frag16 a, b0, b1;
#pragma unroll
    for (int i = 0; i < 8; ++i) {
      a.u[i]     = arow[i];        // K = k0 + kh*16 + i     (16B run)
      a.u[8 + i] = arow[8 + i];    // K = k0 + kh*16 + 8 + i (16B run)
    }
#pragma unroll
    for (int i = 0; i < 16; ++i) {
      b0.u[i] = brow0[i];          // K = k0 + kh*16 + i     (32B run)
      b1.u[i] = brow1[i];
    }
    acc0 = __builtin_amdgcn_wmma_f32_16x16x32_bf16(
        false, a.v, false, b0.v, (short)0, acc0, false, false);
    acc1 = __builtin_amdgcn_wmma_f32_16x16x32_bf16(
        false, a.v, false, b1.v, (short)0, acc1, false, false);
    arow  += 32;
    brow0 += 32;
    brow1 += 32;
  }

  float* crow0 = C + (size_t)(row0 + kh * 8) * N + col0 + n;
#pragma unroll
  for (int r = 0; r < 8; ++r) {
    crow0[(size_t)r * N]      = acc0[r];
    crow0[(size_t)r * N + 16] = acc1[r];
  }
}

// --------------- split wavgk outputs + activations to bf16 -----------------
__global__ __launch_bounds__(576) void split_wavgk_kernel(
    const float* __restrict__ xwavgk,
    unsigned short* __restrict__ xw_t, unsigned short* __restrict__ xa,
    unsigned short* __restrict__ xv,   unsigned short* __restrict__ xg,
    unsigned short* __restrict__ xk) {
  const int row = blockIdx.x;
  const int c   = threadIdx.x;
  float v = xwavgk[(size_t)row * WAVGK + c];
  if (c < 96)       xw_t[(size_t)row * 96  + c]         = f32_to_bf16(tanhf(v));
  else if (c < 192) xa  [(size_t)row * 96  + (c - 96)]  = f32_to_bf16(v);
  else if (c < 256) xv  [(size_t)row * 64  + (c - 192)] = f32_to_bf16(v);
  else if (c < 512) xg  [(size_t)row * 256 + (c - 256)] = f32_to_bf16(sigmoidf_(v));
  else              xk  [(size_t)row * 64  + (c - 512)] = f32_to_bf16(v);
}

// ---- per (b,t,h): biases, head RMSNorms, RoPE, v/k mixing, gates ----------
__global__ __launch_bounds__(64) void prep_kernel(
    const float* __restrict__ rkv,
    const float* __restrict__ wpre, const float* __restrict__ apre,
    const float* __restrict__ vpre, const float* __restrict__ kpre,
    const float* __restrict__ w0,  const float* __restrict__ a0,
    const float* __restrict__ v0,  const float* __restrict__ k0,
    const float* __restrict__ R_bias, const float* __restrict__ K_bias,
    const float* __restrict__ V_bias,
    const float* __restrict__ ln_r, const float* __restrict__ ln_k,
    const float* __restrict__ cosb, const float* __restrict__ sinb,
    const float* __restrict__ v_first, const float* __restrict__ k_first,
    float* __restrict__ r_out, float* __restrict__ k_out,
    float* __restrict__ v_out, float* __restrict__ gk_out,
    float* __restrict__ aa_out, float* __restrict__ bb_out) {
  const int h = blockIdx.x, t = blockIdx.y, b = blockIdx.z, i = threadIdx.x;
  const int hk = h >> 2;                       // kv_repeat = H/HKV = 4
  const size_t bt = (size_t)b * Tc + t;
  __shared__ float red[64];
  __shared__ float buf[64];

  const float cs = cosb[bt * Nn + i];
  const float sn = sinb[bt * Nn + i];

  // r: bias -> per-head RMSNorm(ln_r) -> RoPE
  float rv = rkv[bt * RKVN + h * Nn + i] + R_bias[h * Nn + i];
  float rsum = block64_sum(rv * rv, red, i);
  float rn = rv * rsqrtf(rsum / 64.0f + 1e-6f) * ln_r[i];
  buf[i] = rn; __syncthreads();
  float rrot = (i < 32) ? -buf[i + 32] : buf[i - 32];
  __syncthreads();
  float rq = rn * cs + rrot * sn;

  // k: bias -> per-kv-head RMSNorm(ln_k) -> RoPE
  float kraw = rkv[bt * RKVN + HNc + hk * Nn + i] + K_bias[hk * Nn + i];
  float ksum = block64_sum(kraw * kraw, red, i);
  float kn = kraw * rsqrtf(ksum / 64.0f + 1e-6f) * ln_k[i];
  buf[i] = kn; __syncthreads();
  float krot = (i < 32) ? -buf[i + 32] : buf[i - 32];
  __syncthreads();
  float kq = kn * cs + krot * sn;

  // v / k first-token mixing
  float vv = rkv[bt * RKVN + HNc + KVD + hk * Nn + i] + V_bias[hk * Nn + i];
  float vsig = sigmoidf_(vpre[bt * KVD + hk * Nn + i] + v0[hk * Nn + i]);
  float vfi  = v_first[(bt * HKVn + hk) * Nn + i];
  vv = vv + (vfi - vv) * vsig;
  float ksig = sigmoidf_(kpre[bt * KVD + hk * Nn + i] + k0[hk * Nn + i]);
  float kfi  = k_first[(bt * HKVn + hk) * Nn + i];
  float kf = kq + (kfi - kq) * ksig;

  // gates
  float av   = sigmoidf_(apre[bt * HNc + h * Nn + i] + a0[h * Nn + i]);
  float z    = wpre[bt * HNc + h * Nn + i] + w0[h * Nn + i];
  float wlog = -softplusf_(-z) - 0.5f;

  // kk = k / max(||k||, 1e-12) per head
  float knorm2 = block64_sum(kf * kf, red, i);
  float kkn = kf / fmaxf(sqrtf(knorm2), 1e-12f);

  float kfin = kf * (1.0f - wlog + av);

  const size_t o = bt * HNc + h * Nn + i;
  r_out[o]  = rq;
  k_out[o]  = kfin;
  v_out[o]  = vv;
  gk_out[o] = -expf(wlog);
  aa_out[o] = -kkn;
  bb_out[o] = kkn * av;
}

// --------- sequential RWKV-7 recurrence: 1 block per (b,h) -----------------
// Lane v owns state column S[:,v] (64 fp32 regs). Per step: stage the five
// 64-vectors + decay through LDS, then 2x 64-MAC passes per lane.
__global__ __launch_bounds__(64) void rwkv_scan_kernel(
    const float* __restrict__ r, const float* __restrict__ gk,
    const float* __restrict__ k, const float* __restrict__ v,
    const float* __restrict__ aa, const float* __restrict__ bb,
    const float* __restrict__ S0, float* __restrict__ o) {
  const int b = blockIdx.x >> 5;      // H = 32
  const int h = blockIdx.x & 31;
  const int vc = threadIdx.x;         // state column 0..63
  __shared__ float sr[64], sk[64], sv[64], sa[64], sb[64], sdec[64];

  float S[64];
#pragma unroll
  for (int kk = 0; kk < 64; ++kk)
    S[kk] = S0[(((size_t)b * Hh + h) * Nn + kk) * Nn + vc];

  for (int t = 0; t < Tc; ++t) {
    const size_t base = ((size_t)b * Tc + t) * HNc + h * Nn;
    sr[vc]   = r [base + vc];
    sk[vc]   = k [base + vc];
    sv[vc]   = v [base + vc];
    sa[vc]   = aa[base + vc];
    sb[vc]   = bb[base + vc];
    sdec[vc] = expf(gk[base + vc]);
    __syncthreads();

    float acc_sa = 0.0f;
#pragma unroll
    for (int kk = 0; kk < 64; ++kk) acc_sa += sa[kk] * S[kk];

    const float vval = sv[vc];
    float out = 0.0f;
#pragma unroll
    for (int kk = 0; kk < 64; ++kk) {
      S[kk] = fmaf(sdec[kk], S[kk], fmaf(sk[kk], vval, sb[kk] * acc_sa));
      out = fmaf(sr[kk], S[kk], out);
    }
    o[base + vc] = out;
    __syncthreads();
  }
}

// --------- bonus term + output gate, produce bf16 (y*g) for O GEMM ---------
__global__ __launch_bounds__(64) void bonus_gate_kernel(
    const float* __restrict__ o, const float* __restrict__ r,
    const float* __restrict__ k, const float* __restrict__ v,
    const float* __restrict__ r_k, const float* __restrict__ g,
    unsigned short* __restrict__ yg_bf) {
  const int h = blockIdx.x, t = blockIdx.y, b = blockIdx.z, i = threadIdx.x;
  __shared__ float red[64];
  const size_t bt  = (size_t)b * Tc + t;
  const size_t off = bt * HNc + h * Nn + i;
  float dot = block64_sum(r[off] * k[off] * r_k[h * Nn + i], red, i);
  float y = o[off] * 0.125f + dot * v[off];        // N^-0.5 = 1/8
  yg_bf[off] = f32_to_bf16(y * g[off]);
}

// ---------------------------------------------------------------------------
extern "C" void kernel_launch(void* const* d_in, const int* in_sizes, int n_in,
                              void* d_out, int out_size, void* d_ws, size_t ws_size,
                              hipStream_t stream) {
  (void)in_sizes; (void)n_in; (void)out_size; (void)ws_size;
  const float* x_in    = (const float*)d_in[0];
  const float* v_first = (const float*)d_in[1];
  const float* k_first = (const float*)d_in[2];
  const float* state   = (const float*)d_in[3];
  const float* ccos    = (const float*)d_in[4];
  const float* csin    = (const float*)d_in[5];
  const float* wavgk1  = (const float*)d_in[6];
  const float* w0      = (const float*)d_in[7];
  const float* w2      = (const float*)d_in[8];
  const float* a0      = (const float*)d_in[9];
  const float* a2      = (const float*)d_in[10];
  const float* v0      = (const float*)d_in[11];
  const float* v2      = (const float*)d_in[12];
  const float* g2      = (const float*)d_in[13];
  const float* k0      = (const float*)d_in[14];
  const float* k2      = (const float*)d_in[15];
  const float* r_k     = (const float*)d_in[16];
  const float* RKV     = (const float*)d_in[17];
  const float* Ow      = (const float*)d_in[18];
  const float* R_bias  = (const float*)d_in[19];
  const float* K_bias  = (const float*)d_in[20];
  const float* V_bias  = (const float*)d_in[21];
  const float* ln_r    = (const float*)d_in[22];
  const float* ln_k    = (const float*)d_in[23];
  const float* ln1     = (const float*)d_in[24];
  const float* ln2     = (const float*)d_in[25];

  char* ws = (char*)d_ws;
  size_t off = 0;
  auto carve = [&](size_t bytes) -> void* {
    void* p = ws + off;
    off += (bytes + 255) & ~(size_t)255;
    return p;
  };

  // bf16 activation buffers (A operands, row-major MxK)
  unsigned short* x_bf  = (unsigned short*)carve((size_t)BT * HNc * 2);
  unsigned short* xw_t  = (unsigned short*)carve((size_t)BT * 96  * 2);
  unsigned short* xa_bf = (unsigned short*)carve((size_t)BT * 96  * 2);
  unsigned short* xv_bf = (unsigned short*)carve((size_t)BT * 64  * 2);
  unsigned short* xg_bf = (unsigned short*)carve((size_t)BT * 256 * 2);
  unsigned short* xk_bf = (unsigned short*)carve((size_t)BT * 64  * 2);
  unsigned short* yg_bf = (unsigned short*)carve((size_t)BT * HNc * 2);
  // bf16 weight buffers, TRANSPOSED to NxK for K-contiguous B fragments
  unsigned short* wavgk1_t = (unsigned short*)carve((size_t)WAVGK * HNc * 2);
  unsigned short* RKV_t    = (unsigned short*)carve((size_t)RKVN  * HNc * 2);
  unsigned short* O_t      = (unsigned short*)carve((size_t)HNc   * HNc * 2);
  unsigned short* w2_t     = (unsigned short*)carve((size_t)HNc * 96  * 2);
  unsigned short* a2_t     = (unsigned short*)carve((size_t)HNc * 96  * 2);
  unsigned short* v2_t     = (unsigned short*)carve((size_t)KVD * 64  * 2);
  unsigned short* k2_t     = (unsigned short*)carve((size_t)KVD * 64  * 2);
  unsigned short* g2_t     = (unsigned short*)carve((size_t)HNc * 256 * 2);
  // fp32 buffers
  float* xwavgk = (float*)carve((size_t)BT * WAVGK * 4);
  float* rkv    = (float*)carve((size_t)BT * RKVN * 4);
  float* wpre   = (float*)carve((size_t)BT * HNc * 4);
  float* apre   = (float*)carve((size_t)BT * HNc * 4);
  float* vpre   = (float*)carve((size_t)BT * KVD * 4);
  float* kpre   = (float*)carve((size_t)BT * KVD * 4);
  float* gpre   = (float*)carve((size_t)BT * HNc * 4);
  float* r_buf  = (float*)carve((size_t)BT * HNc * 4);
  float* k_buf  = (float*)carve((size_t)BT * HNc * 4);
  float* v_buf  = (float*)carve((size_t)BT * HNc * 4);
  float* gk_buf = (float*)carve((size_t)BT * HNc * 4);
  float* aa_buf = (float*)carve((size_t)BT * HNc * 4);
  float* bb_buf = (float*)carve((size_t)BT * HNc * 4);
  float* o_buf  = (float*)carve((size_t)BT * HNc * 4);
  float* out2   = (float*)carve((size_t)BT * HNc * 4);

  auto cvt_t = [&](const float* src, unsigned short* dst, int K, int N) {
    size_t total = (size_t)K * N;
    int blocks = (int)((total + 256 * 8 - 1) / (256 * 8));
    cvt_transpose_bf16_kernel<<<blocks, 256, 0, stream>>>(src, dst, K, N);
  };
  auto gemm = [&](const unsigned short* A, const unsigned short* Bt, float* C,
                  int M, int N, int K) {
    dim3 grid(N / 64, M / 16);
    wmma_gemm_bf16<<<grid, 64, 0, stream>>>(A, Bt, C, M, N, K);
  };

  // 1. weights -> bf16, transposed (NxK)
  cvt_t(wavgk1, wavgk1_t, HNc, WAVGK);
  cvt_t(RKV,    RKV_t,    HNc, RKVN);
  cvt_t(Ow,     O_t,      HNc, HNc);
  cvt_t(w2,     w2_t,     96,  HNc);
  cvt_t(a2,     a2_t,     96,  HNc);
  cvt_t(v2,     v2_t,     64,  KVD);
  cvt_t(k2,     k2_t,     64,  KVD);
  cvt_t(g2,     g2_t,     256, HNc);

  // 2. x = rms_norm(x_in, ln1) -> bf16
  rmsnorm_kernel<<<BT, 256, 0, stream>>>(x_in, nullptr, ln1, nullptr, x_bf, HNc);

  // 3. big projections
  gemm(x_bf, wavgk1_t, xwavgk, BT, WAVGK, HNc);
  gemm(x_bf, RKV_t,    rkv,    BT, RKVN,  HNc);

  // 4. split + activations
  split_wavgk_kernel<<<BT, WAVGK, 0, stream>>>(xwavgk, xw_t, xa_bf, xv_bf, xg_bf, xk_bf);

  // 5. second-level projections
  gemm(xw_t,  w2_t, wpre, BT, HNc, 96);
  gemm(xa_bf, a2_t, apre, BT, HNc, 96);
  gemm(xv_bf, v2_t, vpre, BT, KVD, 64);
  gemm(xk_bf, k2_t, kpre, BT, KVD, 64);
  gemm(xg_bf, g2_t, gpre, BT, HNc, 256);

  // 6. per-(b,t,h) preprocessing
  {
    dim3 grid(Hh, Tc, Bc);
    prep_kernel<<<grid, Nn, 0, stream>>>(
        rkv, wpre, apre, vpre, kpre, w0, a0, v0, k0,
        R_bias, K_bias, V_bias, ln_r, ln_k, ccos, csin,
        v_first, k_first, r_buf, k_buf, v_buf, gk_buf, aa_buf, bb_buf);
  }

  // 7. sequential recurrence: one block per (b,h)
  rwkv_scan_kernel<<<Bc * Hh, Nn, 0, stream>>>(
      r_buf, gk_buf, k_buf, v_buf, aa_buf, bb_buf, state, o_buf);

  // 8. bonus + output gate -> bf16 (y*g)
  {
    dim3 grid(Hh, Tc, Bc);
    bonus_gate_kernel<<<grid, Nn, 0, stream>>>(
        o_buf, r_buf, k_buf, v_buf, r_k, gpre, yg_bf);
  }

  // 9. output projection
  gemm(yg_bf, O_t, out2, BT, HNc, HNc);

  // 10. x_new = x_in + out ; d_out = rms_norm(x_new, ln2)
  rmsnorm_kernel<<<BT, 256, 0, stream>>>(x_in, out2, ln2, (float*)d_out, nullptr, HNc);
}
